// NanoMambaCriticalDetector_75385265980113
// MI455X (gfx1250) — compile-verified, matrix operand order checked by reading, not run on hardware
//
#include <hip/hip_runtime.h>
#include <hip/hip_bf16.h>
#include <math.h>

// ---- problem constants ----
#define Bn    64
#define IMG   384
#define PS    16
#define DM    96
#define DS    16
#define DI    96
#define DT_RANK 6
#define NREP  4
#define NCLS  5
#define HPd   (IMG/PS)          // 24
#define Ntok  (HPd*HPd)         // 576
#define Mrows (Bn*Ntok)         // 36864
#define Kpe   (3*PS*PS)         // 768

typedef __bf16 bf16;
typedef __attribute__((ext_vector_type(16))) __bf16 v16bf;
typedef __attribute__((ext_vector_type(8)))  float  v8f;

__device__ __forceinline__ float sigmoidf_(float v) { return 1.0f / (1.0f + __expf(-v)); }
__device__ __forceinline__ float siluf_(float v)    { return v * sigmoidf_(v); }

// Load one 16-bit WMMA fragment (ISA 7.12.2): for lane half hf, the 16 bf16
// values are K = k0+8*hf .. +7 (VGPR0-3) and K = k0+16+8*hf .. +7 (VGPR4-7),
// i.e. two aligned 16-byte blocks of a contiguous bf16 row.
__device__ __forceinline__ v16bf load_frag16(const bf16* __restrict__ row, int k0, int hf) {
    union { v16bf v; uint4 u[2]; } fr;
    fr.u[0] = *(const uint4*)(row + k0 + 8 * hf);
    fr.u[1] = *(const uint4*)(row + k0 + 16 + 8 * hf);
    return fr.v;
}

// -------------------------------------------------------------------------
// K1: 5x5 mean blur (zero-pad SAME) + log1p normalize, written bf16 in im2col
// layout: xn[row=(b,ph,pw)][k=(c,py,px)], row stride Kpe (pure permutation,
// patches don't overlap).
// -------------------------------------------------------------------------
__global__ void blur_im2col_kernel(const float* __restrict__ x, bf16* __restrict__ xn) {
    long idx = (long)blockIdx.x * blockDim.x + threadIdx.x;
    const long total = (long)Bn * 3 * IMG * IMG;
    if (idx >= total) return;
    int b  = (int)(idx / (3L * IMG * IMG));
    int r  = (int)(idx % (3L * IMG * IMG));
    int c  = r / (IMG * IMG);
    int p  = r % (IMG * IMG);
    int yy = p / IMG, xx = p % IMG;
    const float* plane = x + ((long)(b * 3 + c)) * IMG * IMG;
    float L = 0.0f;
    #pragma unroll
    for (int dy = -2; dy <= 2; ++dy) {
        int y2 = yy + dy;
        if (y2 < 0 || y2 >= IMG) continue;
        #pragma unroll
        for (int dx = -2; dx <= 2; ++dx) {
            int x2 = xx + dx;
            if (x2 < 0 || x2 >= IMG) continue;
            L += plane[y2 * IMG + x2];
        }
    }
    L *= (1.0f / 25.0f);
    float xv = plane[yy * IMG + xx];
    float o = log1pf(fmaxf(xv, 0.0f)) - log1pf(fmaxf(L, 0.0f));
    int ph = yy >> 4, py = yy & 15, pw = xx >> 4, px = xx & 15;
    long row = (long)b * Ntok + ph * HPd + pw;
    int  k   = c * 256 + py * 16 + px;
    xn[row * Kpe + k] = (bf16)o;
}

// ---- weight prep: f32 -> bf16, optionally transposing KxNC -> NCxK ----
__global__ void wprep_kernel(const float* __restrict__ W, bf16* __restrict__ Wb, long n) {
    long i = (long)blockIdx.x * blockDim.x + threadIdx.x;
    if (i < n) Wb[i] = (bf16)W[i];
}
__global__ void wprep_t_kernel(const float* __restrict__ W, bf16* __restrict__ Wt,
                               int K, int NC) {   // W: K x NC  ->  Wt: NC x K
    long i = (long)blockIdx.x * blockDim.x + threadIdx.x;
    if (i >= (long)K * NC) return;
    int k = (int)(i % K);
    int n = (int)(i / K);
    Wt[i] = (bf16)W[(size_t)k * NC + n];
}

// -------------------------------------------------------------------------
// Pure-bf16 WMMA GEMM, one wave per 16-row x NC-col strip:
//   C[M x NC] (f32) = act(A[M x K] @ Wt^T + bias)
//   A:  bf16 row-major, row stride K (16B-aligned rows).
//   Wt: bf16 NC x K row-major (pre-transposed weights) -> B[k][n] = Wt[n*K+k].
// Inner loop is loads + v_wmma only: no VALU writes fragment registers, so no
// WMMA->VALU WAR hazard NOPs (VMEM ordering handled by s_wait_loadcnt).
// -------------------------------------------------------------------------
template<int NC, int ACT>
__global__ void gemm_bf16_wmma_kernel(const bf16* __restrict__ A, const bf16* __restrict__ Wt,
                                      const float* __restrict__ bias, float* __restrict__ C,
                                      int M, int K)
{
    constexpr int TN = NC / 16;
    int wave = blockIdx.x * (blockDim.x >> 5) + (threadIdx.x >> 5);
    if (wave >= (M >> 4)) return;              // wave-uniform (EXEC all-1 at WMMA)
    int lane = threadIdx.x & 31;
    int hf   = lane >> 4;                      // lane half (0/1)
    int idx  = lane & 15;                      // M-index for A frag, N-index for B/C

    const bf16* arow = A + (size_t)(wave * 16 + idx) * K;

    v8f acc[TN];
    #pragma unroll
    for (int t = 0; t < TN; ++t) acc[t] = (v8f){};

    for (int k0 = 0; k0 < K; k0 += 32) {
        __builtin_prefetch(arow + k0 + 32, 0, 1);   // -> global_prefetch_b8
        v16bf af = load_frag16(arow, k0, hf);
        #pragma unroll
        for (int t = 0; t < TN; ++t) {
            v16bf bfr = load_frag16(Wt + (size_t)(t * 16 + idx) * K, k0, hf);
            acc[t] = __builtin_amdgcn_wmma_f32_16x16x32_bf16(
                         false, af, false, bfr, (short)0, acc[t], false, false);
        }
    }
    // D layout: VGPR r, lanes 0-15 -> (M=r, N=lane); lanes 16-31 -> (M=8+r, N=lane-16)
    #pragma unroll
    for (int t = 0; t < TN; ++t) {
        int n = t * 16 + idx;
        float bv = bias ? bias[n] : 0.0f;
        #pragma unroll
        for (int r = 0; r < 8; ++r) {
            int m = wave * 16 + r + 8 * hf;
            float v = acc[t][r] + bv;
            if (ACT == 1) v = sigmoidf_(v);
            C[(size_t)m * NC + n] = v;
        }
    }
}

// ---- LayerNorm over last dim C; dual output (f32 and/or bf16), in place OK ----
__global__ void ln_kernel(const float* __restrict__ in, float* __restrict__ outf,
                          bf16* __restrict__ outb,
                          const float* __restrict__ g, const float* __restrict__ b,
                          int M, int C) {
    int row = blockIdx.x * blockDim.x + threadIdx.x;
    if (row >= M) return;
    const float* p = in + (size_t)row * C;
    float m = 0.f;
    for (int c = 0; c < C; ++c) m += p[c];
    m /= C;
    float v = 0.f;
    for (int c = 0; c < C; ++c) { float d = p[c] - m; v += d * d; }
    v /= C;
    float inv = rsqrtf(v + 1e-5f);
    for (int c = 0; c < C; ++c) {
        float r = (p[c] - m) * inv * g[c] + b[c];
        if (outf) outf[(size_t)row * C + c] = r;
        if (outb) outb[(size_t)row * C + c] = (bf16)r;
    }
}

// ---- vis = sigmoid(tokens @ W_vis + b_vis), per row ----
__global__ void vis_kernel(const float* __restrict__ tok, const float* __restrict__ Wv,
                           const float* __restrict__ bv, float* __restrict__ vis, int M) {
    int row = blockIdx.x * blockDim.x + threadIdx.x;
    if (row >= M) return;
    const float* p = tok + (size_t)row * DM;
    float s = bv[0];
    for (int c = 0; c < DM; ++c) s += p[c] * Wv[c];
    vis[row] = sigmoidf_(s);
}

// ---- causal depthwise conv (k=3, pad_left=2) along t, + SiLU ----
__global__ void conv_silu_kernel(const float* __restrict__ xz, const float* __restrict__ cw,
                                 float* __restrict__ xs) {
    long idx = (long)blockIdx.x * blockDim.x + threadIdx.x;
    if (idx >= (long)Mrows * DI) return;
    int d = (int)(idx % DI);
    long row = idx / DI;
    int t = (int)(row % Ntok);
    long b0 = row - t;                        // start row of this batch
    float s = 0.f;
    #pragma unroll
    for (int j = 0; j < 3; ++j) {
        int tt = t - 2 + j;
        if (tt >= 0) s += xz[(b0 + tt) * (2 * DI) + d] * cw[d * 3 + j];
    }
    xs[row * DI + d] = siluf_(s);
}

// ---- proj = xs @ W_xproj  (96 -> 38), scalar ----
__global__ void xproj_kernel(const float* __restrict__ xs, const float* __restrict__ W,
                             float* __restrict__ proj) {
    long idx = (long)blockIdx.x * blockDim.x + threadIdx.x;
    const int O = DT_RANK + 2 * DS;           // 38
    if (idx >= (long)Mrows * O) return;
    int o = (int)(idx % O);
    long row = idx / O;
    const float* p = xs + row * DI;
    float s = 0.f;
    for (int c = 0; c < DI; ++c) s += p[c] * W[c * O + o];
    proj[row * O + o] = s;
}

// ---- dt = softplus(dt_raw @ W_dt + b_dt)  (6 -> 96) ----
__global__ void dt_kernel(const float* __restrict__ proj, const float* __restrict__ Wdt,
                          const float* __restrict__ bdt, float* __restrict__ dt) {
    long idx = (long)blockIdx.x * blockDim.x + threadIdx.x;
    if (idx >= (long)Mrows * DI) return;
    int d = (int)(idx % DI);
    long row = idx / DI;
    const float* p = proj + row * (DT_RANK + 2 * DS);
    float s = bdt[d];
    #pragma unroll
    for (int r = 0; r < DT_RANK; ++r) s += p[r] * Wdt[r * DI + d];
    dt[row * DI + d] = (s > 20.0f) ? s : log1pf(__expf(s));
}

// ---- selective scan: one lane per (b, d); 16 states in registers.
//      Output y is bf16 (feeds only the W_out GEMM). ----
__global__ void scan_kernel(const float* __restrict__ dt, const float* __restrict__ xs,
                            const float* __restrict__ xz, const float* __restrict__ proj,
                            const float* __restrict__ A_log, const float* __restrict__ Dp,
                            const float* __restrict__ vis, bf16* __restrict__ y) {
    int b = blockIdx.x / 3;
    int d = (blockIdx.x % 3) * 32 + (threadIdx.x & 31);
    float a_[DS], hst[DS];
    #pragma unroll
    for (int s = 0; s < DS; ++s) { a_[s] = -__expf(A_log[d * DS + s]); hst[s] = 0.f; }
    float Dd = Dp[d];
    for (int t = 0; t < Ntok; ++t) {
        long row = (long)b * Ntok + t;
        float dt_t = dt[row * DI + d];
        float x_t  = xs[row * DI + d];
        const float* pr = proj + row * (DT_RANK + 2 * DS);
        float acc = 0.f;
        float dx = dt_t * x_t;
        #pragma unroll
        for (int s = 0; s < DS; ++s) {
            float dA = __expf(dt_t * a_[s]);
            hst[s] = dA * hst[s] + dx * pr[DT_RANK + s];
            acc   += hst[s] * pr[DT_RANK + DS + s];
        }
        float z  = xz[row * (2 * DI) + DI + d];
        y[row * DI + d] = (bf16)((acc + Dd * x_t) * siluf_(z) * vis[row]);
    }
}

// ---- tokens += (y @ W_out) * norm_gate ----
__global__ void resid_update_kernel(float* __restrict__ tok, const float* __restrict__ yW,
                                    const float* __restrict__ ng) {
    long idx = (long)blockIdx.x * blockDim.x + threadIdx.x;
    if (idx >= (long)Mrows * DM) return;
    tok[idx] += yW[idx] * ng[idx];
}

// ---- head: depthwise 3x3 conv + BN + SiLU -> gbuf[(b,p),c] ----
__global__ void head_dw_kernel(const float* __restrict__ tok, const float* __restrict__ dw,
                               const float* __restrict__ bng, const float* __restrict__ bnb,
                               const float* __restrict__ bnm, const float* __restrict__ bnv,
                               float* __restrict__ g) {
    long idx = (long)blockIdx.x * blockDim.x + threadIdx.x;
    if (idx >= (long)Mrows * DM) return;
    int c = (int)(idx % DM);
    long row = idx / DM;
    int p = (int)(row % Ntok);
    long b0 = row - p;
    int yy = p / HPd, xx = p % HPd;
    float s = 0.f;
    #pragma unroll
    for (int dy = -1; dy <= 1; ++dy) {
        int y2 = yy + dy; if (y2 < 0 || y2 >= HPd) continue;
        #pragma unroll
        for (int dx = -1; dx <= 1; ++dx) {
            int x2 = xx + dx; if (x2 < 0 || x2 >= HPd) continue;
            s += tok[(b0 + y2 * HPd + x2) * DM + c] * dw[c * 9 + (dy + 1) * 3 + (dx + 1)];
        }
    }
    s = (s - bnm[c]) * rsqrtf(bnv[c] + 1e-5f);
    g[idx] = siluf_(s * bng[c] + bnb[c]);
}

// ---- head 1x1 projections: out[(b,o),p] = feat[(b,p),:] . W[o,:] + bias[o] ----
__global__ void head_proj_kernel(const float* __restrict__ feat, const float* __restrict__ W,
                                 const float* __restrict__ bias, float* __restrict__ out, int O) {
    long idx = (long)blockIdx.x * blockDim.x + threadIdx.x;
    if (idx >= (long)Bn * O * Ntok) return;
    int p = (int)(idx % Ntok);
    int o = (int)((idx / Ntok) % O);
    int b = (int)(idx / ((long)O * Ntok));
    const float* f = feat + ((long)b * Ntok + p) * DM;
    float s = bias[o];
    for (int c = 0; c < DM; ++c) s += f[c] * W[o * DM + c];
    out[idx] = s;
}

// -------------------------------------------------------------------------
static inline int cdiv(long a, int b) { return (int)((a + b - 1) / b); }

extern "C" void kernel_launch(void* const* d_in, const int* in_sizes, int n_in,
                              void* d_out, int out_size, void* d_ws, size_t ws_size,
                              hipStream_t stream) {
    const float* x      = (const float*)d_in[0];
    const float* W_pe   = (const float*)d_in[1];
    const float* b_pe   = (const float*)d_in[2];
    const float* W_vis  = (const float*)d_in[3];
    const float* b_vis  = (const float*)d_in[4];
    const float* dn_g   = (const float*)d_in[5];
    const float* dn_b   = (const float*)d_in[6];
    const float* W_gate = (const float*)d_in[7];
    const float* b_gate = (const float*)d_in[8];
    const float* ln_g   = (const float*)d_in[9];
    const float* ln_b   = (const float*)d_in[10];
    const float* W_in   = (const float*)d_in[11];
    const float* conv_w = (const float*)d_in[12];
    const float* W_xprj = (const float*)d_in[13];
    const float* W_dt   = (const float*)d_in[14];
    const float* b_dt   = (const float*)d_in[15];
    const float* A_log  = (const float*)d_in[16];
    const float* Dp     = (const float*)d_in[17];
    const float* W_out  = (const float*)d_in[18];
    const float* hd_dw  = (const float*)d_in[19];
    const float* bn_g   = (const float*)d_in[20];
    const float* bn_b   = (const float*)d_in[21];
    const float* bn_m   = (const float*)d_in[22];
    const float* bn_v   = (const float*)d_in[23];
    const float* W_heat = (const float*)d_in[24];
    const float* b_heat = (const float*)d_in[25];
    const float* W_off  = (const float*)d_in[26];
    const float* b_off  = (const float*)d_in[27];
    const float* W_size = (const float*)d_in[28];
    const float* b_size = (const float*)d_in[29];

    // ---- workspace layout (float units; bf16 regions use even float counts,
    //      every region starts 16B-aligned) ----
    float* ws = (float*)d_ws;
    size_t o = 0;
    auto alloc_f = [&](size_t nfloats) { float* p = ws + o; o += (nfloats + 3) & ~(size_t)3; return p; };
    float* tokens = alloc_f((size_t)Mrows * DM);
    float* vis    = alloc_f((size_t)Mrows);
    float* ngate  = alloc_f((size_t)Mrows * DM);
    float* xzb    = alloc_f((size_t)Mrows * 2 * DI);
    float* xsb    = alloc_f((size_t)Mrows * DI);
    float* projb  = alloc_f((size_t)Mrows * (DT_RANK + 2 * DS));
    float* dtb    = alloc_f((size_t)Mrows * DI);
    float* yWb    = alloc_f((size_t)Mrows * DM);
    float* gbuf   = alloc_f((size_t)Mrows * DM);
    auto alloc_b = [&](size_t nbf) { bf16* p = (bf16*)(ws + o); o += (nbf / 2 + 3) & ~(size_t)3; return p; };
    bf16* xnb  = alloc_b((size_t)Mrows * Kpe);     // im2col activations (bf16)
    bf16* tokb = alloc_b((size_t)Mrows * DM);      // bf16 shadow of LN(dn) tokens
    bf16* hb   = alloc_b((size_t)Mrows * DM);      // LN output (GEMM input only)
    bf16* ybb  = alloc_b((size_t)Mrows * DI);      // scan output (GEMM input only)
    bf16* wpeb = alloc_b((size_t)DM * Kpe);        // W_pe  (already NCxK)
    bf16* wgt  = alloc_b((size_t)DM * DM);         // W_gate^T
    bf16* wint = alloc_b((size_t)(2 * DI) * DM);   // W_in^T
    bf16* wott = alloc_b((size_t)DM * DM);         // W_out^T

    const int T = 256;
    const int gblocks = (Mrows / 16) / 4;          // 4 waves/block, 16 rows/wave

    // 0) weight prep (bf16, transposed to NCxK)
    wprep_kernel  <<<cdiv((long)DM * Kpe, T), T, 0, stream>>>(W_pe, wpeb, (long)DM * Kpe);
    wprep_t_kernel<<<cdiv((long)DM * DM, T), T, 0, stream>>>(W_gate, wgt, DM, DM);
    wprep_t_kernel<<<cdiv((long)DM * 2 * DI, T), T, 0, stream>>>(W_in, wint, DM, 2 * DI);
    wprep_t_kernel<<<cdiv((long)DM * DM, T), T, 0, stream>>>(W_out, wott, DM, DM);

    // 1) blur + log-normalize + im2col (bf16)
    blur_im2col_kernel<<<cdiv((long)Bn * 3 * IMG * IMG, T), T, 0, stream>>>(x, xnb);

    // 2) patch embed GEMM: tokens = xn @ W_pe^T + b_pe   (M=36864, N=96, K=768)
    gemm_bf16_wmma_kernel<DM, 0><<<gblocks, 128, 0, stream>>>(xnb, wpeb, b_pe, tokens, Mrows, Kpe);
    // 3) vis from pre-LN tokens
    vis_kernel<<<cdiv(Mrows, T), T, 0, stream>>>(tokens, W_vis, b_vis, vis, Mrows);
    // 4) tokens = LN(tokens, dn) in place (+ bf16 shadow for the gate GEMM)
    ln_kernel<<<cdiv(Mrows, T), T, 0, stream>>>(tokens, tokens, tokb, dn_g, dn_b, Mrows, DM);
    // 5) norm_gate = sigmoid(tokens @ W_gate + b_gate)
    gemm_bf16_wmma_kernel<DM, 1><<<gblocks, 128, 0, stream>>>(tokb, wgt, b_gate, ngate, Mrows, DM);

    // 6) Mamba blocks
    for (int rep = 0; rep < NREP; ++rep) {
        ln_kernel<<<cdiv(Mrows, T), T, 0, stream>>>(tokens, nullptr, hb, ln_g, ln_b, Mrows, DM);
        // xz = h @ W_in  (96 -> 192)
        gemm_bf16_wmma_kernel<2 * DI, 0><<<gblocks, 128, 0, stream>>>(hb, wint, nullptr, xzb, Mrows, DM);
        conv_silu_kernel<<<cdiv((long)Mrows * DI, T), T, 0, stream>>>(xzb, conv_w, xsb);
        xproj_kernel<<<cdiv((long)Mrows * (DT_RANK + 2 * DS), T), T, 0, stream>>>(xsb, W_xprj, projb);
        dt_kernel<<<cdiv((long)Mrows * DI, T), T, 0, stream>>>(projb, W_dt, b_dt, dtb);
        scan_kernel<<<Bn * 3, 32, 0, stream>>>(dtb, xsb, xzb, projb, A_log, Dp, vis, ybb);
        // yW = y @ W_out
        gemm_bf16_wmma_kernel<DM, 0><<<gblocks, 128, 0, stream>>>(ybb, wott, nullptr, yWb, Mrows, DM);
        resid_update_kernel<<<cdiv((long)Mrows * DM, T), T, 0, stream>>>(tokens, yWb, ngate);
    }

    // 7) head
    head_dw_kernel<<<cdiv((long)Mrows * DM, T), T, 0, stream>>>(tokens, hd_dw, bn_g, bn_b,
                                                                bn_m, bn_v, gbuf);
    float* out  = (float*)d_out;
    float* heat = out;
    float* offp = out + (size_t)Bn * NCLS * Ntok;
    float* szp  = offp + (size_t)Bn * 2 * Ntok;
    head_proj_kernel<<<cdiv((long)Bn * NCLS * Ntok, T), T, 0, stream>>>(gbuf, W_heat, b_heat, heat, NCLS);
    head_proj_kernel<<<cdiv((long)Bn * 2 * Ntok, T), T, 0, stream>>>(tokens, W_off,  b_off,  offp, 2);
    head_proj_kernel<<<cdiv((long)Bn * 2 * Ntok, T), T, 0, stream>>>(tokens, W_size, b_size, szp, 2);
}
